// GD_1030792151437
// MI455X (gfx1250) — matrix operand order repaired and checked by reading, not compiled, not run on hardware
//
#include <hip/hip_runtime.h>
#include <hip/hip_bf16.h>

typedef unsigned short     u16;
typedef unsigned int       u32;
typedef unsigned long long u64;
typedef __attribute__((ext_vector_type(16))) __bf16 bf16x16;
typedef __attribute__((ext_vector_type(8)))  float  f32x8;

struct __align__(16) U4 { u32 x, y, z, w; };
union Frag { bf16x16 v; U4 q[2]; };

static constexpr int B  = 2, S = 1024, V = 32000, D = 512, H = 8, L = 2;
static constexpr int FF = 2048;          // 4*D
static constexpr int NE = 528;           // D padded to 33*16; row D of wte_extT = ones

__device__ inline u16 f2bf(float f) {
  union { float f; u32 u; } c; c.f = f;
  u32 u = c.u;
  return (u16)((u + 0x7FFFu + ((u >> 16) & 1u)) >> 16);   // RNE f32->bf16
}

// ---------- block reductions (256 threads, wave32) ----------
__device__ inline float blk_sum(float v, float* sb) {
  int t = threadIdx.x;
  #pragma unroll
  for (int o = 16; o; o >>= 1) v += __shfl_down(v, o, 32);
  if ((t & 31) == 0) sb[t >> 5] = v;
  __syncthreads();
  if (t == 0) { float s = 0.f; for (int i = 0; i < 8; ++i) s += sb[i]; sb[0] = s; }
  __syncthreads();
  float r = sb[0];
  __syncthreads();
  return r;
}
__device__ inline float blk_max(float v, float* sb) {
  int t = threadIdx.x;
  #pragma unroll
  for (int o = 16; o; o >>= 1) v = fmaxf(v, __shfl_down(v, o, 32));
  if ((t & 31) == 0) sb[t >> 5] = v;
  __syncthreads();
  if (t == 0) { float s = sb[0]; for (int i = 1; i < 8; ++i) s = fmaxf(s, sb[i]); sb[0] = s; }
  __syncthreads();
  float r = sb[0];
  __syncthreads();
  return r;
}

// ---------- WMMA GEMM (single layout: C = A·Bᵀ) -----------------------------
// A bf16 (M,K) row-major, B bf16 (N,K) row-major, C f32 (M,N).
// 128x128 block tile, BK=32, 8 waves (2x4), wave tile 64x32 (4x2 WMMA tiles).
// Full tiles: 3-stage circular LDS pipeline fed by GLOBAL_LOAD_ASYNC_TO_LDS_B128
// (ASYNCcnt-tracked), 2-tile prefetch depth, ONE barrier per K-step.
#define BM 128
#define BN 128
#define BK 32
#define LDT 40   // LDS row stride in halfwords (80B = 20 banks -> conflict-free)

#define WAITA4 asm volatile("s_wait_asynccnt 0x4" ::: "memory")
#define WAITA0 asm volatile("s_wait_asynccnt 0x0" ::: "memory")

__device__ inline void async_b128(u32 lds_off, u64 gaddr) {
  asm volatile("global_load_async_to_lds_b128 %0, %1, off"
               :: "v"(lds_off), "v"(gaddr) : "memory");
}

__device__ inline void load_frags(const u16* As, const u16* Bs,
                                  int wm, int wn, int lo, int hi,
                                  Frag a[4], Frag b[2]) {
  #pragma unroll
  for (int ii = 0; ii < 4; ++ii) {
    // A 16x32: lane lo = M; elem e -> K = (e&7) + 8*hi + 16*(e>>3)
    const u16* p = As + (wm + ii * 16 + lo) * LDT;
    a[ii].q[0] = *(const U4*)(p + 8 * hi);
    a[ii].q[1] = *(const U4*)(p + 16 + 8 * hi);
  }
  #pragma unroll
  for (int jj = 0; jj < 2; ++jj) {
    // B 32x16: lane lo = N; elem e -> K = (e&7) + 8*hi + 16*(e>>3)
    const u16* p = Bs + (wn + jj * 16 + lo) * LDT;
    b[jj].q[0] = *(const U4*)(p + 8 * hi);
    b[jj].q[1] = *(const U4*)(p + 16 + 8 * hi);
  }
}

template<int ACCUM>
__global__ void __launch_bounds__(256)
gemm_wmma(const u16* A, const u16* Bp, float* C,
          int Mm, int Nn, int Kk, int lda, int ldb, int ldc)
{
  __shared__ __align__(16) u16 As[3][BM * LDT];
  __shared__ __align__(16) u16 Bs[3][BN * LDT];
  const int t    = threadIdx.x;
  const int m0   = blockIdx.x * BM;
  const int n0   = blockIdx.y * BN;
  const int lane = t & 31;
  const int lo   = lane & 15, hi = lane >> 4;
  const int w    = t >> 5;
  const int wm   = (w >> 2) * 64;   // 0 / 64
  const int wn   = (w & 3) * 32;    // 0 / 32 / 64 / 96

  const f32x8 fz = {0.f, 0.f, 0.f, 0.f, 0.f, 0.f, 0.f, 0.f};
  f32x8 acc[4][2];
  #pragma unroll
  for (int i = 0; i < 4; ++i)
    #pragma unroll
    for (int j = 0; j < 2; ++j) acc[i][j] = fz;

  if (m0 + BM <= Mm && n0 + BN <= Nn) {
    // ============ fast path: async 3-stage pipeline, 1 barrier/step ============
    u32 asb[3] = { (u32)(size_t)&As[0][0], (u32)(size_t)&As[1][0], (u32)(size_t)&As[2][0] };
    u32 bsb[3] = { (u32)(size_t)&Bs[0][0], (u32)(size_t)&Bs[1][0], (u32)(size_t)&Bs[2][0] };
    auto issue = [&](int k0, int buf) {
      #pragma unroll
      for (int i = 0; i < 2; ++i) {
        int c = t + i * 256, row = c >> 2, ch = (c & 3) << 3;
        u32 loff = (u32)((row * LDT + ch) * 2);
        async_b128(asb[buf] + loff, (u64)(A  + (size_t)(m0 + row) * lda + k0 + ch));
        async_b128(bsb[buf] + loff, (u64)(Bp + (size_t)(n0 + row) * ldb + k0 + ch));
      }
    };
    auto compute = [&](int buf) {
      Frag a[4], b[2];
      load_frags(&As[buf][0], &Bs[buf][0], wm, wn, lo, hi, a, b);
      #pragma unroll
      for (int ii = 0; ii < 4; ++ii)
        #pragma unroll
        for (int jj = 0; jj < 2; ++jj)
          acc[ii][jj] = __builtin_amdgcn_wmma_f32_16x16x32_bf16(
              false, a[ii].v, false, b[jj].v, (short)0, acc[ii][jj], false, false);
    };
    const int nt = Kk / BK;          // >= 16 for every GEMM here
    issue(0, 0);
    issue(BK, 1);
    int cur = 0, i = 0;
    for (; i + 2 < nt; ++i) {        // branch-free steady state
      WAITA4;                         // tile i landed (in-order completion)
      __syncthreads();                // publish; also fences compute(i-1) vs issue(i+2)
      compute(cur);
      issue((i + 2) * BK, (cur + 2) % 3);
      cur = (cur == 2) ? 0 : cur + 1;
    }
    WAITA4;  __syncthreads();  compute(cur);          // tile nt-2
    cur = (cur == 2) ? 0 : cur + 1;
    WAITA0;  __syncthreads();  compute(cur);          // tile nt-1
    // unguarded store
    #pragma unroll
    for (int ii = 0; ii < 4; ++ii)
      #pragma unroll
      for (int jj = 0; jj < 2; ++jj)
        #pragma unroll
        for (int vv = 0; vv < 8; ++vv) {
          size_t idx = (size_t)(m0 + wm + ii * 16 + vv + 8 * hi) * ldc
                     + (n0 + wn + jj * 16 + lo);
          if (ACCUM) C[idx] += acc[ii][jj][vv];
          else       C[idx]  = acc[ii][jj][vv];
        }
  } else {
    // ================= edge path: guarded, single buffer =================
    for (int k0 = 0; k0 < Kk; k0 += BK) {
      __syncthreads();
      #pragma unroll
      for (int i = 0; i < 2; ++i) {
        int cid = t + i * 256, row = cid >> 2, ch = (cid & 3) << 3;
        U4 va = {0u, 0u, 0u, 0u}, vb = {0u, 0u, 0u, 0u};
        if (m0 + row < Mm) va = *(const U4*)(A  + (size_t)(m0 + row) * lda + k0 + ch);
        if (n0 + row < Nn) vb = *(const U4*)(Bp + (size_t)(n0 + row) * ldb + k0 + ch);
        *(U4*)(&As[0][0] + row * LDT + ch) = va;
        *(U4*)(&Bs[0][0] + row * LDT + ch) = vb;
      }
      __syncthreads();
      Frag a[4], b[2];
      load_frags(&As[0][0], &Bs[0][0], wm, wn, lo, hi, a, b);
      #pragma unroll
      for (int ii = 0; ii < 4; ++ii)
        #pragma unroll
        for (int jj = 0; jj < 2; ++jj)
          acc[ii][jj] = __builtin_amdgcn_wmma_f32_16x16x32_bf16(
              false, a[ii].v, false, b[jj].v, (short)0, acc[ii][jj], false, false);
    }
    #pragma unroll
    for (int ii = 0; ii < 4; ++ii)
      #pragma unroll
      for (int jj = 0; jj < 2; ++jj)
        #pragma unroll
        for (int vv = 0; vv < 8; ++vv) {
          int m = m0 + wm + ii * 16 + vv + 8 * hi;
          int n = n0 + wn + jj * 16 + lo;
          if (m < Mm && n < Nn) {
            size_t idx = (size_t)m * ldc + n;
            if (ACCUM) C[idx] += acc[ii][jj][vv];
            else       C[idx]  = acc[ii][jj][vv];
          }
        }
  }
}

// ---------- elementwise / reduction kernels ---------------------------------
__global__ void __launch_bounds__(256)
k_fill(float* p, float v, int n) {
  int i = blockIdx.x * 256 + threadIdx.x;
  if (i < n) p[i] = v;
}

__global__ void __launch_bounds__(256)
k_cvt(const float* in, u16* out, int n) {
  int i = blockIdx.x * 256 + threadIdx.x;
  if (i < n) out[i] = f2bf(in[i]);
}

__global__ void __launch_bounds__(256)
k_gelu(const float* in, u16* out, int n) {
  int i = blockIdx.x * 256 + threadIdx.x;
  if (i < n) {
    float x = in[i];
    out[i] = f2bf(0.5f * x * (1.f + erff(x * 0.70710678118654752f)));
  }
}

// rows x D layernorm; optional int gather of input rows; f32 and/or bf16 out
__global__ void __launch_bounds__(256)
k_ln_rows(const float* in, const int* gather, const float* w,
          float* out_f, u16* out_b, int rows) {
  __shared__ float sb[8];
  int row = blockIdx.x;
  if (row >= rows) return;
  size_t src = gather ? (size_t)gather[row] : (size_t)row;
  const float* p = in + src * D;
  int t = threadIdx.x;
  float x0 = p[t], x1 = p[t + 256];
  float mu = blk_sum(x0 + x1, sb) * (1.f / D);
  float d0 = x0 - mu, d1 = x1 - mu;
  float var = blk_sum(d0 * d0 + d1 * d1, sb) * (1.f / D);
  float inv = rsqrtf(var + 1e-5f);
  float y0 = d0 * inv * w[t], y1 = d1 * inv * w[t + 256];
  size_t o = (size_t)row * D;
  if (out_f) { out_f[o + t] = y0; out_f[o + t + 256] = y1; }
  if (out_b) { out_b[o + t] = f2bf(y0); out_b[o + t + 256] = f2bf(y1); }
}

// p = LN(wpe[r]); K[r] = p*Wk (r<S); Q[r-1] = p*Wq (r>=1)
__global__ void __launch_bounds__(256)
k_pq(const float* wpe, const float* lnw, const float* Wq, const float* Wk,
     u16* Qb, u16* Kb) {
  __shared__ float sb[8];
  int r = blockIdx.x, t = threadIdx.x;          // r in [0, S]
  const float* p = wpe + (size_t)r * D;
  float x0 = p[t], x1 = p[t + 256];
  float mu = blk_sum(x0 + x1, sb) * (1.f / D);
  float d0 = x0 - mu, d1 = x1 - mu;
  float var = blk_sum(d0 * d0 + d1 * d1, sb) * (1.f / D);
  float inv = rsqrtf(var + 1e-5f);
  float y0 = d0 * inv * lnw[t], y1 = d1 * inv * lnw[t + 256];
  if (r < S) {
    Kb[(size_t)r * D + t]       = f2bf(y0 * Wk[t]);
    Kb[(size_t)r * D + t + 256] = f2bf(y1 * Wk[t + 256]);
  }
  if (r >= 1) {
    Qb[(size_t)(r - 1) * D + t]       = f2bf(y0 * Wq[t]);
    Qb[(size_t)(r - 1) * D + t + 256] = f2bf(y1 * Wq[t + 256]);
  }
}

// causal softmax row s of scores (S x S); folds 1/sqrt(D) and N_reg = 1/(s+1)
__global__ void __launch_bounds__(256)
k_softmax_causal(const float* scores, u16* krn) {
  __shared__ float sb[8];
  int s = blockIdx.x, t = threadIdx.x;
  const float scale = 0.04419417382415922f;     // 1/sqrt(512)
  float r[4]; float mx = -3.0e38f;
  #pragma unroll
  for (int i = 0; i < 4; ++i) {
    int j = t + i * 256;
    float v = (j <= s) ? scores[(size_t)s * S + j] * scale : -3.0e38f;
    r[i] = v; mx = fmaxf(mx, v);
  }
  mx = blk_max(mx, sb);
  float sum = 0.f;
  #pragma unroll
  for (int i = 0; i < 4; ++i) {
    int j = t + i * 256;
    float e = (j <= s) ? expf(r[i] - mx) : 0.f;
    r[i] = e; sum += e;
  }
  sum = blk_sum(sum, sb);
  float inv = 1.f / (sum * (float)(s + 1));
  #pragma unroll
  for (int i = 0; i < 4; ++i)
    krn[(size_t)s * S + t + i * 256] = f2bf(r[i] * inv);
}

// softmax down the columns of lT (S x V): one column per thread, bf16 out
__global__ void __launch_bounds__(256)
k_colsoftmax(const float* lT, u16* outT) {
  int v = blockIdx.x * 256 + threadIdx.x;       // < V
  float m = -3.0e38f;
  for (int s = 0; s < S; ++s) m = fmaxf(m, lT[(size_t)s * V + v]);
  float sum = 0.f;
  for (int s = 0; s < S; ++s) sum += expf(lT[(size_t)s * V + v] - m);
  float inv = 1.f / sum;
  for (int s = 0; s < S; ++s)
    outT[(size_t)s * V + v] = f2bf(expf(lT[(size_t)s * V + v] - m) * inv);
}

// column mean of wte (V x D) -> cs[D]   (cs pre-zeroed)
__global__ void __launch_bounds__(256)
k_colmean(const float* wte, float* cs) {
  int v0 = blockIdx.x * 64, t = threadIdx.x;
  float a0 = 0.f, a1 = 0.f;
  for (int i = 0; i < 64; ++i) {
    const float* r = wte + (size_t)(v0 + i) * D;
    a0 += r[2 * t]; a1 += r[2 * t + 1];
  }
  atomicAdd(&cs[2 * t],     a0 * (1.f / V));
  atomicAdd(&cs[2 * t + 1], a1 * (1.f / V));
}

// wte_extT (NE, V) bf16: rows [0,D) = wteᵀ, row D = 1.0, rest 0
__global__ void __launch_bounds__(256)
k_wte_extT(const float* wte, u16* out) {
  size_t i = (size_t)blockIdx.x * 256 + threadIdx.x;
  if (i >= (size_t)NE * V) return;
  int n = (int)(i / V); size_t v = i % V;
  u16 r = 0;
  if (n < D)       r = f2bf(wte[v * D + n]);
  else if (n == D) r = 0x3F80;                   // bf16(1.0)
  out[i] = r;
}

// W_eff[k][d][dp] = sum_h W_o[k][d][h*D+dp]  -> bf16 (L, D, D)
__global__ void __launch_bounds__(256)
k_weff(const float* W_o, u16* out, int n) {
  int i = blockIdx.x * 256 + threadIdx.x;
  if (i >= n) return;
  int dp = i % D, d = (i / D) % D, k = i / (D * D);
  const float* base = W_o + ((size_t)k * D + d) * (size_t)(H * D) + dp;
  float s = 0.f;
  #pragma unroll
  for (int h = 0; h < H; ++h) s += base[h * D];
  out[i] = f2bf(s);
}

// Vvᵀ (b, D, S) bf16 via LDS-tiled transpose.
// Vv = (e - avg)*Wv ; avg = cmean[d] (numv==null) or num[row*NE+d]/num[row*NE+D]
__global__ void __launch_bounds__(256)
k_vvT(const float* e, const float* avg, const float* numv, const float* Wv,
      u16* outT) {
  __shared__ u16 tile[32][33];
  int b = blockIdx.z;
  int s0 = blockIdx.x * 32, d0 = blockIdx.y * 32;
  int td = threadIdx.x & 31, ts = threadIdx.x >> 5;       // ts in [0,8)
  #pragma unroll
  for (int i = 0; i < 4; ++i) {
    int s = s0 + ts + i * 8, d = d0 + td;
    size_t row = (size_t)b * S + s;
    float a = numv ? numv[row * NE + d] / numv[row * NE + D] : avg[d];
    tile[ts + i * 8][td] = f2bf((e[row * D + d] - a) * Wv[d]);
  }
  __syncthreads();
  #pragma unroll
  for (int i = 0; i < 4; ++i) {
    int d = d0 + ts + i * 8, s = s0 + td;
    outT[((size_t)b * D + d) * S + s] = tile[td][ts + i * 8];
  }
}

// cross-entropy over rows of logits (B*S rows of length V); mean via atomics
__global__ void __launch_bounds__(256)
k_loss(const float* logits, const int* tgt, float* loss) {
  __shared__ float sb[8];
  size_t row = blockIdx.x;
  const float* p = logits + row * (size_t)V;
  float mx = -3.0e38f;
  for (int j = threadIdx.x; j < V; j += 256) mx = fmaxf(mx, p[j]);
  mx = blk_max(mx, sb);
  float s = 0.f;
  for (int j = threadIdx.x; j < V; j += 256) s += expf(p[j] - mx);
  s = blk_sum(s, sb);
  if (threadIdx.x == 0) {
    float lse = mx + logf(s);
    atomicAdd(loss, -(p[tgt[row]] - lse) * (1.f / (B * S)));
  }
}

// ---------------------------------------------------------------------------
extern "C" void kernel_launch(void* const* d_in, const int* in_sizes, int n_in,
                              void* d_out, int out_size, void* d_ws, size_t ws_size,
                              hipStream_t stream) {
  (void)in_sizes; (void)n_in; (void)out_size; (void)ws_size;
  const int*   x      = (const int*)d_in[0];
  const int*   tgt    = (const int*)d_in[1];
  const float* wte    = (const float*)d_in[2];
  const float* wpe    = (const float*)d_in[3];
  const float* ln_e_w = (const float*)d_in[4];
  const float* ln_p_w = (const float*)d_in[5];
  const float* Wq     = (const float*)d_in[6];
  const float* Wk     = (const float*)d_in[7];
  const float* Wv     = (const float*)d_in[8];
  const float* W_o    = (const float*)d_in[9];
  const float* ff_ln  = (const float*)d_in[10];
  const float* ff_w1  = (const float*)d_in[11];
  const float* ff_w2  = (const float*)d_in[12];
  const float* ln_o_w = (const float*)d_in[13];

  char* ws = (char*)d_ws;
  size_t off = 0;
  auto alloc = [&](size_t bytes) -> void* {
    void* p = ws + off;
    off = (off + bytes + 255) & ~(size_t)255;
    return p;
  };
  float* e     = (float*)alloc((size_t)B * S * D * 4);
  u16*   Qb    = (u16*)  alloc((size_t)S * D * 2);
  u16*   Kb    = (u16*)  alloc((size_t)S * D * 2);
  float* sc    = (float*)alloc((size_t)S * S * 4);
  u16*   krn   = (u16*)  alloc((size_t)S * S * 2);
  u16*   wteb  = (u16*)  alloc((size_t)V * D * 2);
  u16*   wteXT = (u16*)  alloc((size_t)NE * V * 2);
  u16*   wteLn = (u16*)  alloc((size_t)V * D * 2);
  float* cmean = (float*)alloc((size_t)D * 4);
  float* fk    = (float*)alloc((size_t)B * S * D * 4);
  u16*   fkb   = (u16*)  alloc((size_t)B * S * D * 2);
  u16*   VvT   = (u16*)  alloc((size_t)B * D * S * 2);
  float* u     = (float*)alloc((size_t)B * S * D * 4);
  u16*   ub    = (u16*)  alloc((size_t)B * S * D * 2);
  u16*   Weff  = (u16*)  alloc((size_t)L * D * D * 2);
  float* num   = (float*)alloc((size_t)B * S * NE * 4);
  float* g     = (float*)alloc((size_t)B * S * FF * 4);
  u16*   h1b   = (u16*)  alloc((size_t)B * S * FF * 2);
  u16*   w1b   = (u16*)  alloc((size_t)FF * D * 2);
  u16*   w2b   = (u16*)  alloc((size_t)D * FF * 2);
  u16*   tb    = (u16*)  alloc((size_t)B * S * D * 2);

  float* logits = (float*)d_out;
  float* lossp  = logits + (size_t)B * S * V;
  float* lT     = logits;                                  // (S,V) f32 scratch
  u16*   RwT    = (u16*)(logits + (size_t)S * V);          // (S,V) bf16 scratch

  dim3 blk(256);
  // ---- weight prep ----
  k_fill<<<1, blk, 0, stream>>>(lossp, 0.f, 1);
  k_fill<<<2, blk, 0, stream>>>(cmean, 0.f, D);
  k_cvt<<<(int)(((size_t)V * D + 255) / 256), blk, 0, stream>>>(wte, wteb, V * D);
  k_wte_extT<<<(int)(((size_t)NE * V + 255) / 256), blk, 0, stream>>>(wte, wteXT);
  k_ln_rows<<<V, blk, 0, stream>>>(wte, nullptr, ln_e_w, nullptr, wteLn, V);
  k_cvt<<<(FF * D + 255) / 256, blk, 0, stream>>>(ff_w1, w1b, FF * D);
  k_cvt<<<(D * FF + 255) / 256, blk, 0, stream>>>(ff_w2, w2b, D * FF);
  k_weff<<<(L * D * D + 255) / 256, blk, 0, stream>>>(W_o, Weff, L * D * D);
  k_colmean<<<V / 64, blk, 0, stream>>>(wte, cmean);

  // ---- embeddings / positions / attention kernel ----
  k_ln_rows<<<B * S, blk, 0, stream>>>(wte, x, ln_e_w, e, nullptr, B * S);
  k_pq<<<S + 1, blk, 0, stream>>>(wpe, ln_p_w, Wq, Wk, Qb, Kb);
  gemm_wmma<0><<<dim3(S / BM, S / BN), blk, 0, stream>>>(Qb, Kb, sc, S, S, D, D, D, S);
  k_softmax_causal<<<S, blk, 0, stream>>>(sc, krn);

  // ---- layer 0 (f_k == 0 -> R uniform -> avg_wte = colmean(wte)) ----
  k_vvT<<<dim3(S / 32, D / 32, B), blk, 0, stream>>>(e, cmean, nullptr, Wv, VvT);
  for (int b = 0; b < B; ++b)
    gemm_wmma<0><<<dim3(S / BM, D / BN), blk, 0, stream>>>(
        krn, VvT + (size_t)b * D * S, u + (size_t)b * S * D, S, D, S, S, S, D);
  k_cvt<<<(B * S * D + 255) / 256, blk, 0, stream>>>(u, ub, B * S * D);
  gemm_wmma<0><<<dim3(B * S / BM, D / BN), blk, 0, stream>>>(ub, Weff, fk, B * S, D, D, D, D, D);

  // ---- layer 1 ----
  k_cvt<<<(B * S * D + 255) / 256, blk, 0, stream>>>(fk, fkb, B * S * D);
  for (int b = 0; b < B; ++b) {
    // lᵀ = f_k · wteᵀ  (S x V, f32, in d_out half 0)
    gemm_wmma<0><<<dim3(S / BM, V / BN), blk, 0, stream>>>(
        fkb + (size_t)b * S * D, wteb, lT, S, V, D, D, D, V);
    // R = softmax over s (columns); bf16 into d_out half 1
    k_colsoftmax<<<V / 256, blk, 0, stream>>>(lT, RwT);
    // [num | denom] = R · [wte | 1]   (S x NE)
    gemm_wmma<0><<<dim3(S / BM, (NE + BN - 1) / BN), blk, 0, stream>>>(
        RwT, wteXT, num + (size_t)b * S * NE, S, NE, V, V, V, NE);
  }
  k_vvT<<<dim3(S / 32, D / 32, B), blk, 0, stream>>>(e, nullptr, num, Wv, VvT);
  for (int b = 0; b < B; ++b)
    gemm_wmma<0><<<dim3(S / BM, D / BN), blk, 0, stream>>>(
        krn, VvT + (size_t)b * D * S, u + (size_t)b * S * D, S, D, S, S, S, D);
  k_cvt<<<(B * S * D + 255) / 256, blk, 0, stream>>>(u, ub, B * S * D);
  gemm_wmma<1><<<dim3(B * S / BM, D / BN), blk, 0, stream>>>(
      ub, Weff + (size_t)D * D, fk, B * S, D, D, D, D, D);

  // ---- FFN ----
  k_ln_rows<<<B * S, blk, 0, stream>>>(fk, nullptr, ff_ln, nullptr, tb, B * S);
  gemm_wmma<0><<<dim3(B * S / BM, FF / BN), blk, 0, stream>>>(tb, w1b, g, B * S, FF, D, D, D, FF);
  k_gelu<<<(B * S * FF + 255) / 256, blk, 0, stream>>>(g, h1b, B * S * FF);
  gemm_wmma<1><<<dim3(B * S / BM, D / BN), blk, 0, stream>>>(h1b, w2b, fk, B * S, D, FF, FF, FF, D);

  // ---- output head ----
  k_ln_rows<<<B * S, blk, 0, stream>>>(fk, nullptr, ln_o_w, nullptr, fkb, B * S);
  gemm_wmma<0><<<dim3(B * S / BM, V / BN), blk, 0, stream>>>(fkb, wteLn, logits, B * S, V, D, D, D, V);
  k_loss<<<B * S, blk, 0, stream>>>(logits, tgt, lossp);
}
